// QConv2d_60344290508884
// MI455X (gfx1250) — compile-verified, matrix operand order checked by reading, not run on hardware
//
#include <hip/hip_runtime.h>

typedef __attribute__((ext_vector_type(16))) _Float16 v16h;
typedef __attribute__((ext_vector_type(8)))  _Float16 v8h;
typedef __attribute__((ext_vector_type(8)))  float    v8f;
typedef __attribute__((ext_vector_type(4)))  unsigned int v4u;
typedef __attribute__((ext_vector_type(8)))  int v8i;
typedef __attribute__((ext_vector_type(4)))  int v4i;

// Problem geometry (fixed by the reference)
constexpr int BATCH = 16;
constexpr int CIN   = 32;
constexpr int HIN   = 56;
constexpr int WIN   = 56;
constexpr int COUT  = 64;
constexpr int HPAD  = 58;            // 56 + 2*padding
constexpr int WPAD  = 58;
constexpr int OHW   = HIN * WIN;     // 3136 output pixels per image
constexpr int MT    = OHW / 16;      // 196 m-tiles per image
constexpr int NT    = COUT / 16;     // 4  n-tiles
constexpr int TAPS  = 9;             // 3x3 kernel -> 9 K-chunks of 32
constexpr int WROWS = TAPS * COUT;   // 576 rows of 32 halves in packed weights

// float_quantize(EXP=5, MAN=10) == fp16 RNE with saturation at +-65504
__device__ __forceinline__ float fq(float v) {
    v = fminf(fmaxf(v, -65504.0f), 65504.0f);
    return (float)(_Float16)v;
}

// ---------------------------------------------------------------------------
// Pack x: NCHW fp32 -> padded NHWC fp16   xt[b][ih][iw][c], pad ring = 0.
// ---------------------------------------------------------------------------
__global__ __launch_bounds__(256) void pack_x(const float* __restrict__ x,
                                              _Float16* __restrict__ xt) {
    int idx = blockIdx.x * 256 + threadIdx.x;
    int c  = idx & (CIN - 1);
    int t  = idx >> 5;
    int iw = t % WPAD; t /= WPAD;
    int ih = t % HPAD; t /= HPAD;
    int b  = t;
    if (b >= BATCH) return;
    float v = 0.0f;
    int h = ih - 1, w = iw - 1;
    if (h >= 0 && h < HIN && w >= 0 && w < WIN)
        v = x[((b * CIN + c) * HIN + h) * WIN + w];
    xt[idx] = (_Float16)v;
}

// ---------------------------------------------------------------------------
// Pack weights: w[n][c][ti][tj] fp32 -> wp[tap][n][c] fp16 (quantized).
// ---------------------------------------------------------------------------
__global__ __launch_bounds__(256) void pack_w(const float* __restrict__ w,
                                              _Float16* __restrict__ wp) {
    int idx = blockIdx.x * 256 + threadIdx.x;
    if (idx >= WROWS * CIN) return;
    int c   = idx & (CIN - 1);
    int t   = idx >> 5;
    int n   = t % COUT;
    int tap = t / COUT;
    int ti  = tap / 3, tj = tap % 3;
    float v = w[((n * CIN + c) * 3 + ti) * 3 + tj];
    v = fminf(fmaxf(v, -65504.0f), 65504.0f);
    wp[idx] = (_Float16)v;
}

__global__ void pack_b(const float* __restrict__ bias, float* __restrict__ qb) {
    int i = threadIdx.x;
    if (i < COUT) qb[i] = fq(bias[i]);
}

// ---------------------------------------------------------------------------
// Fused im2col GEMM.
//   * one TENSOR_LOAD_TO_LDS DMA per workgroup stages all packed weights
//   * one wave per 16-pixel m-tile computes all 4 n-tiles (N=64)
//   * 9 taps x 4 n-tiles = 36 x v_wmma_f32_16x16x32_f16 per wave
// ---------------------------------------------------------------------------
__global__ __launch_bounds__(256) void qconv_wmma(const _Float16* __restrict__ xt,
                                                  const _Float16* __restrict__ wp,
                                                  const float* __restrict__ qb,
                                                  float* __restrict__ out) {
    __shared__ _Float16 bsm[WROWS * CIN];          // 36,864 B of LDS

    // ---- TDM: DMA the whole packed-weight tensor into LDS (wave 0 only) ----
    if (threadIdx.x < 32) {
        unsigned lds_off = (unsigned)(uintptr_t)(&bsm[0]);
        unsigned long long ga = (unsigned long long)(uintptr_t)wp;
        // D# group0: count=1 | lds_addr | global_addr[56:0] | type=2
        v4u g0 = { 1u, lds_off, (unsigned)ga,
                   (unsigned)((ga >> 32) & 0x01ffffffull) | 0x80000000u };
        // D# group1: mask=0, data_size=1(2B), tensor_dim0=32, tensor_dim1=576,
        //            tile_dim0=32, tile_dim1=576, tensor_dim0_stride=32
        v8i g1 = { 0x10000,                // [17:16] data_size = 1 (2 bytes)
                   (int)(32u << 16),       // [79:48] tensor_dim0 = 32 (low half)
                   (int)(576u << 16),      // [111:80] tensor_dim1 = 576 (low half)
                   (int)(32u << 16),       // [127:112] tile_dim0 = 32
                   576,                    // [143:128] tile_dim1 = 576
                   32,                     // [207:160] tensor_dim0_stride = 32
                   0, 0 };
        v4i g2 = { 1, 1, 0, 0 };           // tensor_dim2=1, tensor_dim3=1 (unused dims)
        v4i g3 = { 0, 1 << 16, 0, 0 };     // tensor_dim4=1 (unused)
        v8i g4 = { 0, 0, 0, 0, 0, 0, 0, 0 };   // extra group slot (clang-23 form)
        __builtin_amdgcn_tensor_load_to_lds(g0, g1, g2, g3, g4, 0);
        __builtin_amdgcn_s_wait_tensorcnt(0);
    }
    __syncthreads();

    int wave = (blockIdx.x * 256 + threadIdx.x) >> 5;   // global m-tile id
    int lane = threadIdx.x & 31;
    int b  = wave / MT;
    int mt = wave % MT;

    int l = lane & 15;                 // A-row / B-column index
    int h = lane >> 4;                 // K-half selector

    int p  = mt * 16 + l;              // output pixel for this A row
    int oh = p / WIN;
    int ow = p % WIN;
    const _Float16* abase = xt + (((size_t)b * HPAD + oh) * WPAD + ow) * CIN + h * 8;

    v8f acc[NT] = {};
    #pragma unroll
    for (int tap = 0; tap < TAPS; ++tap) {
        int ti = tap / 3, tj = tap % 3;
        const _Float16* ap = abase + ((size_t)ti * WPAD + tj) * CIN;
        v8h alo = *(const v8h*)(ap);        // K = 8h .. 8h+7
        v8h ahi = *(const v8h*)(ap + 16);   // K = 16+8h .. 23+8h
        v16h A;
        #pragma unroll
        for (int i = 0; i < 8; ++i) { A[i] = alo[i]; A[i + 8] = ahi[i]; }

        #pragma unroll
        for (int nt = 0; nt < NT; ++nt) {
            const _Float16* bp = &bsm[((size_t)(tap * COUT + nt * 16 + l)) * CIN + h * 8];
            v8h blo = *(const v8h*)(bp);    // ds_load_b128
            v8h bhi = *(const v8h*)(bp + 16);
            v16h B;
            #pragma unroll
            for (int i = 0; i < 8; ++i) { B[i] = blo[i]; B[i + 8] = bhi[i]; }
            acc[nt] = __builtin_amdgcn_wmma_f32_16x16x32_f16(
                false, A, false, B, (short)0, acc[nt], false, false);
        }
    }

    // C/D layout: acc[nt][r] = (M = r + 8h, N = nt*16 + l). Output NCHW fp32.
    #pragma unroll
    for (int nt = 0; nt < NT; ++nt) {
        int n = nt * 16 + l;
        float bv = qb[n];
        float* obase = out + ((size_t)(b * COUT + n)) * OHW + mt * 16 + h * 8;
        #pragma unroll
        for (int r = 0; r < 8; ++r)
            obase[r] = fq(acc[nt][r] + bv);   // 8 contiguous floats -> 2x b128 store
    }
}

// ---------------------------------------------------------------------------
extern "C" void kernel_launch(void* const* d_in, const int* in_sizes, int n_in,
                              void* d_out, int out_size, void* d_ws, size_t ws_size,
                              hipStream_t stream) {
    const float* x    = (const float*)d_in[0];
    const float* w    = (const float*)d_in[1];
    const float* bias = (const float*)d_in[2];
    float* out = (float*)d_out;

    const size_t XT_ELEMS = (size_t)BATCH * HPAD * WPAD * CIN;    // 1,722,368 halves
    const size_t WP_ELEMS = (size_t)WROWS * CIN;                  // 18,432 halves
    _Float16* xt = (_Float16*)d_ws;
    _Float16* wp = xt + XT_ELEMS;
    float*    qb = (float*)(wp + WP_ELEMS);

    pack_x<<<(int)((XT_ELEMS + 255) / 256), 256, 0, stream>>>(x, xt);
    pack_w<<<(int)((WP_ELEMS + 255) / 256), 256, 0, stream>>>(w, wp);
    pack_b<<<1, 64, 0, stream>>>(bias, qb);

    int mtiles = BATCH * MT;              // 3136 wave-tiles, 8 waves/block
    qconv_wmma<<<mtiles / 8, 256, 0, stream>>>(xt, wp, qb, out);
}
